// LPNET_20109036880229
// MI455X (gfx1250) — compile-verified, matrix operand order checked by reading, not run on hardware
//
#include <hip/hip_runtime.h>
#include <math.h>

// ---------------------------------------------------------------------------
// MI455X (gfx1250) ResNet-18 + head, compile-time-shaped implicit-GEMM WMMA.
// wave32; 256-thread blocks = 8 waves; block tile 128M x 64N; each wave owns a
// 2x2 grid of 16x16 f32 accumulators and issues 8 v_wmma per 64-wide K chunk.
// ---------------------------------------------------------------------------

typedef __attribute__((ext_vector_type(16))) _Float16 v16h;
typedef __attribute__((ext_vector_type(8)))  _Float16 v8h;
typedef __attribute__((ext_vector_type(8)))  float    v8f;

#define KC 64                 // K chunk (two 16x16x32 WMMA k-substeps)
#define MT 128                // M tile per block (4 wave-rows x 32)
#define NT 64                 // N tile per block (2 wave-cols x 32)
#define LS (KC + 8)           // LDS row stride (144 B: keeps 16 B alignment)

__device__ __forceinline__ v8f wmma16(v16h a, v16h b, v8f c) {
    return __builtin_amdgcn_wmma_f32_16x16x32_f16(
        /*neg_a=*/false, a, /*neg_b=*/false, b,
        /*c_mod=*/(short)0, c, /*reuse_a=*/false, /*reuse_b=*/false);
}

// Assemble one 16-bit A/B fragment per ISA layout from LDS:
// lanes 0-15 hold K{0..7,16..23}; lanes 16-31 hold K{8..15,24..31}
__device__ __forceinline__ v16h ldfrag(const _Float16* p) {
    v8h lo = *(const v8h*)p;
    v8h hi = *(const v8h*)(p + 16);
    return __builtin_shufflevector(lo, hi, 0,1,2,3,4,5,6,7,8,9,10,11,12,13,14,15);
}

// ---------------------------------------------------------------------------
// conv + folded BN (+residual)(+ReLU), implicit GEMM, fully constexpr-shaped:
//   out[n,m] = BN( sum_k A[m,k]*W[n,k] ) (+res[n,m]) (ReLU)
//   k = (c*KH + kh)*KW + kw ; m = oh*Wout + ow  (all divisors compile-time)
// Grid: x = ceil(M/128), y = Cout/64.  Block: 256 threads.
// ---------------------------------------------------------------------------
template <int Cin, int Hin, int Win, int Cout, int Hout, int Wout,
          int KH, int KW, int S, int P, bool RELU, bool HASRES>
__global__ __launch_bounds__(256) void conv_k(
    const float* __restrict__ in, const float* __restrict__ wt,
    const float* __restrict__ bng, const float* __restrict__ bnb,
    const float* __restrict__ bnm, const float* __restrict__ bnv,
    const float* __restrict__ res, float* __restrict__ out)
{
    constexpr int K    = Cin * KH * KW;
    constexpr int M    = Hout * Wout;
    constexpr int KHKW = KH * KW;

    __shared__ _Float16 As[MT][LS];
    __shared__ _Float16 Bs[NT][LS];

    const int mBase = blockIdx.x * MT;
    const int nBase = blockIdx.y * NT;
    const int tid   = threadIdx.x;
    const int lane  = tid & 31;
    const int l     = lane & 15;
    const int half  = lane >> 4;
    const int wid   = tid >> 5;
    const int wmi   = (wid & 3) << 5;    // wave M offset (0,32,64,96)
    const int wni   = (wid >> 2) << 5;   // wave N offset (0,32)

    v8f acc[2][2] = {};

    constexpr int Ktiles = (K + KC - 1) / KC;
    for (int kt = 0; kt < Ktiles; ++kt) {
        const int kBase = kt * KC;

        // ---- stage A (im2col) tile: 128 x 64 fp32 -> f16 ----
        for (int i = tid; i < MT * KC; i += 256) {
            int row = i >> 6, kk = i & 63;
            int k = kBase + kk, mm = mBase + row;
            bool ok = true;
            if constexpr (K % KC != 0) ok = ok && (k < K);
            if constexpr (M % MT != 0) ok = ok && (mm < M);
            float val = 0.f;
            if (ok) {
                int c, kh, kw;
                if constexpr (KH == 1 && KW == 1) { c = k; kh = 0; kw = 0; }
                else {
                    c = k / KHKW; int rr = k - c * KHKW;  // div by constant
                    kh = rr / KW; kw = rr - kh * KW;
                }
                int oh = mm / Wout, ow = mm - oh * Wout;  // div by constant
                int ih = oh * S - P + kh;
                int iw = ow * S - P + kw;
                if constexpr (P > 0) {
                    if ((unsigned)ih < (unsigned)Hin && (unsigned)iw < (unsigned)Win)
                        val = in[((size_t)c * Hin + ih) * Win + iw];
                } else {
                    val = in[((size_t)c * Hin + ih) * Win + iw];  // always in-bounds
                }
            }
            As[row][kk] = (_Float16)val;
        }
        // ---- stage B (weights) tile: 64 x 64 (OIHW row n contiguous in k) ----
        for (int i = tid; i < NT * KC; i += 256) {
            int col = i >> 6, kk = i & 63;
            int k = kBase + kk;
            float val = 0.f;
            if constexpr (K % KC != 0) {
                if (k < K) val = wt[(size_t)(nBase + col) * K + k];
            } else {
                val = wt[(size_t)(nBase + col) * K + k];
            }
            Bs[col][kk] = (_Float16)val;
        }
        __syncthreads();

        if (kt + 1 < Ktiles)  // pull next weight K-chunk (global_prefetch_b8)
            __builtin_prefetch(&wt[(size_t)(nBase + (tid >> 2)) * K + kBase + KC], 0, 1);

        // ---- 2x2 register tile: 8 WMMAs per wave per chunk ----
#pragma unroll
        for (int ks = 0; ks < KC; ks += 32) {
            v16h a0 = ldfrag(&As[wmi +  0 + l][ks + half * 8]);
            v16h a1 = ldfrag(&As[wmi + 16 + l][ks + half * 8]);
            v16h b0 = ldfrag(&Bs[wni +  0 + l][ks + half * 8]);
            v16h b1 = ldfrag(&Bs[wni + 16 + l][ks + half * 8]);
            acc[0][0] = wmma16(a0, b0, acc[0][0]);
            acc[0][1] = wmma16(a0, b1, acc[0][1]);
            acc[1][0] = wmma16(a1, b0, acc[1][0]);
            acc[1][1] = wmma16(a1, b1, acc[1][1]);
        }
        __syncthreads();
    }

    // ---- epilogue: folded BN (+res)(+ReLU); C/D: M = r + 8*half, N = l ----
#pragma unroll
    for (int ni = 0; ni < 2; ++ni) {
        const int n = nBase + wni + ni * 16 + l;
        const float sc = bng[n] * rsqrtf(bnv[n] + 1e-5f);
        const float bs = bnb[n] - bnm[n] * sc;
#pragma unroll
        for (int mi = 0; mi < 2; ++mi) {
            const int mb = mBase + wmi + mi * 16 + (half << 3);  // 8 consecutive m
            v8f a = acc[mi][ni];
            if constexpr (M % MT == 0) {
                float o[8];
                if constexpr (HASRES) {
                    float4 r0 = *(const float4*)&res[(size_t)n * M + mb];
                    float4 r1 = *(const float4*)&res[(size_t)n * M + mb + 4];
                    o[0] = r0.x; o[1] = r0.y; o[2] = r0.z; o[3] = r0.w;
                    o[4] = r1.x; o[5] = r1.y; o[6] = r1.z; o[7] = r1.w;
                } else {
                    for (int r = 0; r < 8; ++r) o[r] = 0.f;
                }
#pragma unroll
                for (int r = 0; r < 8; ++r) {
                    float val = a[r] * sc + bs + o[r];
                    if constexpr (RELU) val = fmaxf(val, 0.f);
                    o[r] = val;
                }
                *(float4*)&out[(size_t)n * M + mb]     = make_float4(o[0], o[1], o[2], o[3]);
                *(float4*)&out[(size_t)n * M + mb + 4] = make_float4(o[4], o[5], o[6], o[7]);
            } else {
#pragma unroll
                for (int r = 0; r < 8; ++r) {
                    int mm = mb + r;
                    if (mm < M) {
                        float val = a[r] * sc + bs;
                        if constexpr (HASRES) val += res[(size_t)n * M + mm];
                        if constexpr (RELU)   val = fmaxf(val, 0.f);
                        out[(size_t)n * M + mm] = val;
                    }
                }
            }
        }
    }
}

// ---------------------------------------------------------------------------
// MaxPool 3x3 s2 pad1: 64x224x224 -> 64x112x112
// ---------------------------------------------------------------------------
__global__ __launch_bounds__(256) void maxpool_k(const float* __restrict__ in,
                                                 float* __restrict__ out)
{
    int idx = blockIdx.x * 256 + threadIdx.x;
    const int Q = 64 * 112 * 112;
    if (idx >= Q) return;
    int c = idx / (112 * 112);
    int r = idx - c * (112 * 112);
    int oh = r / 112, ow = r - oh * 112;
    float mx = -INFINITY;
#pragma unroll
    for (int kh = 0; kh < 3; ++kh)
#pragma unroll
        for (int kw = 0; kw < 3; ++kw) {
            int ih = oh * 2 - 1 + kh, iw = ow * 2 - 1 + kw;
            if ((unsigned)ih < 224u && (unsigned)iw < 224u)
                mx = fmaxf(mx, in[((size_t)c * 224 + ih) * 224 + iw]);
        }
    out[idx] = mx;
}

// ---------------------------------------------------------------------------
// Adaptive avg pool: 512 x 14 x 14 -> feat[512]
// ---------------------------------------------------------------------------
__global__ __launch_bounds__(256) void avgpool_k(const float* __restrict__ in,
                                                 float* __restrict__ feat)
{
    int c = blockIdx.x * 256 + threadIdx.x;
    if (c >= 512) return;
    float s = 0.f;
    for (int i = 0; i < 196; ++i) s += in[(size_t)c * 196 + i];
    feat[c] = s * (1.0f / 196.0f);
}

// ---------------------------------------------------------------------------
// Head: f0 (256x528)+ReLU, f1 (128x272)+ReLU, then the collapsed GRU scan.
// h0 == 0 each step => gh = b_hh; only h_new[0:2] consumed => only w_ih rows
// {0,1,128,129,256,257}; dynamic dot maintained incrementally (2 MAC/row/step).
// ---------------------------------------------------------------------------
__device__ __forceinline__ float sigf(float x) { return 1.0f / (1.0f + expf(-x)); }

__global__ __launch_bounds__(256) void head_k(
    const float* __restrict__ feat, const float* __restrict__ bp,
    const float* __restrict__ f0w, const float* __restrict__ f0b,
    const float* __restrict__ f1w, const float* __restrict__ f1b,
    const float* __restrict__ wih, const float* __restrict__ bih,
    const float* __restrict__ bhh, float* __restrict__ out)
{
    __shared__ float x0[528];
    __shared__ float p1x[272];
    __shared__ float p2c[144];
    __shared__ float wdyn[6][64];
    __shared__ float red[6][32];
    __shared__ float gs[6];

    const int t = threadIdx.x;
    const int wid = t >> 5, lane = t & 31;

    for (int i = t; i < 512; i += 256) x0[i] = feat[i];
    if (t < 16) x0[512 + t] = bp[t];
    __syncthreads();

    {   // f0
        float s = f0b[t];
        for (int k = 0; k < 528; ++k) s += f0w[(size_t)t * 528 + k] * x0[k];
        p1x[t] = fmaxf(s, 0.f);
    }
    if (t < 16) p1x[256 + t] = bp[t];
    __syncthreads();

    if (t < 128) {  // f1
        float s = f1b[t];
        for (int k = 0; k < 272; ++k) s += f1w[(size_t)t * 272 + k] * p1x[k];
        p2c[t] = fmaxf(s, 0.f);
    }
    if (t < 16) p2c[128 + t] = bp[t];

    for (int i = t; i < 6 * 64; i += 256) {
        int j = i >> 6, k = i & 63;
        int rowR = (j >> 1) * 128 + (j & 1);
        wdyn[j][k] = wih[(size_t)rowR * 208 + 144 + k];
    }
    __syncthreads();

    if (wid < 6) {
        int rowR = (wid >> 1) * 128 + (wid & 1);
        float s = 0.f;
        for (int k = lane; k < 144; k += 32) s += wih[(size_t)rowR * 208 + k] * p2c[k];
        red[wid][lane] = s;
    }
    __syncthreads();
    for (int off = 16; off > 0; off >>= 1) {
        if (wid < 6 && lane < off) red[wid][lane] += red[wid][lane + off];
        __syncthreads();
    }
    if (t < 6) {
        int rowR = (t >> 1) * 128 + (t & 1);
        gs[t] = red[t][0] + bih[rowR];
    }
    __syncthreads();

    if (t == 0) {
        float a0 = 0.f, a1 = 0.f, a2 = 0.f, a3 = 0.f, a4 = 0.f, a5 = 0.f;
        const float ghr0 = bhh[0],   ghr1 = bhh[1];
        const float ghz0 = bhh[128], ghz1 = bhh[129];
        const float ghn0 = bhh[256], ghn1 = bhh[257];
        for (int i = 0; i < 32; ++i) {
            float r0 = sigf(gs[0] + a0 + ghr0);
            float r1 = sigf(gs[1] + a1 + ghr1);
            float z0 = sigf(gs[2] + a2 + ghz0);
            float z1 = sigf(gs[3] + a3 + ghz1);
            float n0 = tanhf(gs[4] + a4 + r0 * ghn0);
            float n1 = tanhf(gs[5] + a5 + r1 * ghn1);
            float h0 = (1.f - z0) * n0;
            float h1 = (1.f - z1) * n1;
            out[2 * i]     = h0;
            out[2 * i + 1] = h1;
            a0 += wdyn[0][2 * i] * h0 + wdyn[0][2 * i + 1] * h1;
            a1 += wdyn[1][2 * i] * h0 + wdyn[1][2 * i + 1] * h1;
            a2 += wdyn[2][2 * i] * h0 + wdyn[2][2 * i + 1] * h1;
            a3 += wdyn[3][2 * i] * h0 + wdyn[3][2 * i + 1] * h1;
            a4 += wdyn[4][2 * i] * h0 + wdyn[4][2 * i + 1] * h1;
            a5 += wdyn[5][2 * i] * h0 + wdyn[5][2 * i + 1] * h1;
        }
    }
}

// ---------------------------------------------------------------------------
// Host-side orchestration
// ---------------------------------------------------------------------------
struct BN  { const float *g, *b, *m, *v; };
struct Blk { const float* c1; BN b1; const float* c2; BN b2;
             const float* dc; BN db; };

template <int Cin, int Hin, int Win, int Cout, int Hout, int Wout,
          int KH, int KW, int S, int P, bool RELU, bool HASRES>
static inline void launch_conv(hipStream_t st, const float* in, const float* wt,
                               const BN& bn, const float* res, float* out)
{
    constexpr int M = Hout * Wout;
    dim3 g((M + MT - 1) / MT, Cout / NT);
    conv_k<Cin, Hin, Win, Cout, Hout, Wout, KH, KW, S, P, RELU, HASRES>
        <<<g, 256, 0, st>>>(in, wt, bn.g, bn.b, bn.m, bn.v, res, out);
}

extern "C" void kernel_launch(void* const* d_in, const int* in_sizes, int n_in,
                              void* d_out, int out_size, void* d_ws, size_t ws_size,
                              hipStream_t stream)
{
    (void)in_sizes; (void)n_in; (void)out_size; (void)ws_size;

    // ---- input leaves, recursive dict-insertion order (110 leaves) ----
    int idx = 0;
    auto F = [&](void) { return (const float*)d_in[idx++]; };

    const float* ie     = F();                    // [1,8,448,448]
    const float* bp     = F();                    // [16]
    const float* stem_w = F();                    // [64,8,7,7]
    BN stem_bn; stem_bn.g = F(); stem_bn.b = F(); stem_bn.m = F(); stem_bn.v = F();

    Blk blk[4][2];
    for (int si = 0; si < 4; ++si)
        for (int bi = 0; bi < 2; ++bi) {
            Blk& B = blk[si][bi];
            B.c1 = F();
            B.b1.g = F(); B.b1.b = F(); B.b1.m = F(); B.b1.v = F();
            B.c2 = F();
            B.b2.g = F(); B.b2.b = F(); B.b2.m = F(); B.b2.v = F();
            if (si > 0 && bi == 0) {
                B.dc = F();
                B.db.g = F(); B.db.b = F(); B.db.m = F(); B.db.v = F();
            } else {
                B.dc = nullptr;
                B.db.g = B.db.b = B.db.m = B.db.v = nullptr;
            }
        }
    const float* f0w = F(); const float* f0b = F();
    const float* f1w = F(); const float* f1b = F();
    const float* wih = F(); const float* whh = F();  (void)whh; // dead: h0==0
    const float* bih = F(); const float* bhh = F();

    // ---- workspace carve (~16 MB): conv1out region == 4 ping buffers ----
    float* wsf = (float*)d_ws;
    const int QSZ = 64 * 112 * 112;               // 802816
    float* conv1out = wsf;                        // 64*224*224 == 4*QSZ
    float* Q0 = conv1out;
    float* Q1 = conv1out + QSZ;
    float* Q2 = conv1out + 2 * QSZ;
    float* Q3 = conv1out + 3 * QSZ;
    float* poolout = conv1out + 4 * QSZ;
    float* feat    = poolout + QSZ;

    // ---- stem ----
    launch_conv<8, 448, 448, 64, 224, 224, 7, 7, 2, 3, true, false>(
        stream, ie, stem_w, stem_bn, nullptr, conv1out);
    maxpool_k<<<(QSZ + 255) / 256, 256, 0, stream>>>(conv1out, poolout);

    // ---- stage 0: 64ch @112 ----
    launch_conv<64,112,112, 64,112,112, 3,3,1,1, true, false>(stream, poolout, blk[0][0].c1, blk[0][0].b1, nullptr, Q0);
    launch_conv<64,112,112, 64,112,112, 3,3,1,1, true, true >(stream, Q0,      blk[0][0].c2, blk[0][0].b2, poolout, Q1);
    launch_conv<64,112,112, 64,112,112, 3,3,1,1, true, false>(stream, Q1,      blk[0][1].c1, blk[0][1].b1, nullptr, Q0);
    launch_conv<64,112,112, 64,112,112, 3,3,1,1, true, true >(stream, Q0,      blk[0][1].c2, blk[0][1].b2, Q1,      Q2);

    // ---- stage 1: 64->128, 112->56 ----
    launch_conv<64,112,112,128, 56, 56, 3,3,2,1, true, false>(stream, Q2, blk[1][0].c1, blk[1][0].b1, nullptr, Q0);
    launch_conv<64,112,112,128, 56, 56, 1,1,2,0, false,false>(stream, Q2, blk[1][0].dc, blk[1][0].db, nullptr, Q1);
    launch_conv<128,56, 56,128, 56, 56, 3,3,1,1, true, true >(stream, Q0, blk[1][0].c2, blk[1][0].b2, Q1,      Q3);
    launch_conv<128,56, 56,128, 56, 56, 3,3,1,1, true, false>(stream, Q3, blk[1][1].c1, blk[1][1].b1, nullptr, Q0);
    launch_conv<128,56, 56,128, 56, 56, 3,3,1,1, true, true >(stream, Q0, blk[1][1].c2, blk[1][1].b2, Q3,      Q1);

    // ---- stage 2: 128->256, 56->28 ----
    launch_conv<128,56, 56,256, 28, 28, 3,3,2,1, true, false>(stream, Q1, blk[2][0].c1, blk[2][0].b1, nullptr, Q0);
    launch_conv<128,56, 56,256, 28, 28, 1,1,2,0, false,false>(stream, Q1, blk[2][0].dc, blk[2][0].db, nullptr, Q2);
    launch_conv<256,28, 28,256, 28, 28, 3,3,1,1, true, true >(stream, Q0, blk[2][0].c2, blk[2][0].b2, Q2,      Q3);
    launch_conv<256,28, 28,256, 28, 28, 3,3,1,1, true, false>(stream, Q3, blk[2][1].c1, blk[2][1].b1, nullptr, Q0);
    launch_conv<256,28, 28,256, 28, 28, 3,3,1,1, true, true >(stream, Q0, blk[2][1].c2, blk[2][1].b2, Q3,      Q1);

    // ---- stage 3: 256->512, 28->14 ----
    launch_conv<256,28, 28,512, 14, 14, 3,3,2,1, true, false>(stream, Q1, blk[3][0].c1, blk[3][0].b1, nullptr, Q0);
    launch_conv<256,28, 28,512, 14, 14, 1,1,2,0, false,false>(stream, Q1, blk[3][0].dc, blk[3][0].db, nullptr, Q2);
    launch_conv<512,14, 14,512, 14, 14, 3,3,1,1, true, true >(stream, Q0, blk[3][0].c2, blk[3][0].b2, Q2,      Q3);
    launch_conv<512,14, 14,512, 14, 14, 3,3,1,1, true, false>(stream, Q3, blk[3][1].c1, blk[3][1].b1, nullptr, Q0);
    launch_conv<512,14, 14,512, 14, 14, 3,3,1,1, true, true >(stream, Q0, blk[3][1].c2, blk[3][1].b2, Q3,      Q1);

    // ---- pool + head ----
    avgpool_k<<<2, 256, 0, stream>>>(Q1, feat);
    head_k<<<1, 256, 0, stream>>>(feat, bp, f0w, f0b, f1w, f1b,
                                  wih, bih, bhh, (float*)d_out);
}
